// LocationPredictor_6382321402527
// MI455X (gfx1250) — matrix operand order repaired and checked by reading, not compiled, not run on hardware
//
#include <hip/hip_runtime.h>
#include <hip/hip_bf16.h>

// ---------------------------------------------------------------------------
// MI455X (gfx1250) implementation.
//
// Dominant cost: 3 masked-conv steps == per-batch GEMM (256x2304)@(2304x16),
// 2048 batches -> ~116 GFLOP.  Inputs total ~4MB -> compute bound, so we run
// it on the WMMA pipe in bf16 (v_wmma_f32_16x16x32_bf16, f32 accumulate).
//
// K is reordered t-major (k = t*256 + i) so every 32-wide K-tile has a fixed
// conv tap; per-action weights (only 4 distinct mask vectors exist) are
// pre-scaled and pre-swizzled into exact WMMA A-fragment lane order in the
// workspace (L2-resident, 4 x 1.15MB).  Activations live in LDS channel-minor
// with a zero halo, so B fragments are two contiguous ds_load_b128.
// ---------------------------------------------------------------------------

typedef __attribute__((ext_vector_type(16))) __bf16 v16bf;
typedef __attribute__((ext_vector_type(8)))  float  v8f;

union FragA { uint4 u[2]; __bf16 h[16]; v16bf v; };

#define E_DIM      256
#define S_DIM      4
#define KT_N       72          // 2304 / 32 K-tiles
#define FRAG_ELEMS 512         // 32 lanes * 16 bf16 per A fragment
#define APACK_BYTES ((size_t)4 * 16 * KT_N * FRAG_ELEMS * 2)   // 4,718,592

// ---------------- weight pack: masks -> per-action swizzled bf16 A tiles ----
__global__ void __launch_bounds__(256)
pack_weights_kernel(const float* __restrict__ conv_weight,
                    const float* __restrict__ action_table,
                    const float* __restrict__ action_W,
                    const float* __restrict__ action_b,
                    __bf16* __restrict__ Apack)
{
  __shared__ float masks[4][9];
  const int tid = threadIdx.x;
  if (tid < 36) {
    int a = tid / 9, c = tid % 9;
    float acc = action_b[c];
    for (int k = 0; k < 32; ++k) acc += action_table[a*32 + k] * action_W[c*32 + k];
    masks[a][c] = acc;
  }
  __syncthreads();
  if (tid < 4) {
    float mx = masks[tid][0];
    for (int c = 1; c < 9; ++c) mx = fmaxf(mx, masks[tid][c]);
    float e[9]; float s = 0.f;
    for (int c = 0; c < 9; ++c) { e[c] = __expf(masks[tid][c] - mx); s += e[c]; }
    for (int c = 0; c < 9; ++c) masks[tid][c] = e[c] / s;
  }
  __syncthreads();

  const int gid  = blockIdx.x * 256 + tid;      // one thread per (fragment, lane)
  const int lane = gid & 31;
  const int frag = gid >> 5;                    // [0, 4*16*72)
  if (frag >= 4 * 16 * KT_N) return;
  const int kt    = frag % KT_N;
  const int mt    = (frag / KT_N) & 15;
  const int a     = frag / (KT_N * 16);
  const int t     = kt >> 3;                    // conv tap 0..8 (fixed per K-tile)
  const int ibase = (kt & 7) * 32;              // input-channel base of this K-tile
  const int M     = mt * 16 + (lane & 15);      // output channel (A row)
  const float scale = masks[a][t];
  const int klo   = (lane < 16) ? 0 : 8;

  // A-fragment lane layout (16-bit A 16x32): lanes 0-15 hold Klocal 0-15;
  // lanes 16-31 hold Klocal 16-31 (matches contiguous B fragment).
  FragA f;
  #pragma unroll
  for (int e2 = 0; e2 < 16; ++e2) {
    int kl = e2 + 2 * klo;                      // K within tile
    int i  = ibase + kl;                        // input channel
    f.h[e2] = (__bf16)(conv_weight[(M * E_DIM + i) * 9 + t] * scale);
  }
  uint4* dst = (uint4*)(Apack + (size_t)frag * FRAG_ELEMS + lane * 16);
  dst[0] = f.u[0];
  dst[1] = f.u[1];
}

// ---------------- per-level logits accumulation -----------------------------
__device__ __forceinline__ void accum_logits(const __bf16* buf, const float* emb,
                                             float* logits, int lvl, int tid)
{
  const int p  = tid >> 4;                       // pixel 0..15
  const int c0 = (tid & 15) * 16;                // channel chunk
  const int sp = ((p >> 2) + 1) * 6 + ((p & 3) + 1);
  const __bf16* cp = buf + sp * E_DIM + c0;
  const float*  ep = emb + lvl * E_DIM + c0;
  float part = 0.f;
  #pragma unroll
  for (int e = 0; e < 16; ++e) part += (float)cp[e] * ep[e];
  atomicAdd(&logits[p], part);                   // ds_add_f32
}

// ---------------- main: one workgroup (8 wave32) per batch element ----------
__global__ void __launch_bounds__(256)
location_predictor_kernel(const float* __restrict__ gold_table,
                          const float* __restrict__ map_table,
                          const int*   __restrict__ goldstandard,
                          const int*   __restrict__ actions,
                          const int*   __restrict__ landmarks,
                          const int*   __restrict__ yv,
                          const __bf16* __restrict__ Apack,
                          float* __restrict__ prob_out,
                          float* __restrict__ loss_b,
                          float* __restrict__ acc_b)
{
  __shared__ alignas(32) __bf16 curpad[2][36 * E_DIM];   // 6x6 halo x 256 ch, x2
  __shared__ float emb_s[S_DIM * E_DIM];
  __shared__ float logits_s[16];
  __shared__ int   lmk_s[128];
  __shared__ int   gs_s[64];

  const int b   = blockIdx.x;
  const int tid = threadIdx.x;

  if (tid < 128) lmk_s[tid] = landmarks[b * 128 + tid];
  if (tid < 64)  gs_s[tid]  = goldstandard[b * 64 + tid];
  if (tid < 16)  logits_s[tid] = 0.f;
  {
    uint32_t* cz = (uint32_t*)&curpad[0][0];
    for (int i = tid; i < (2 * 36 * E_DIM) / 2; i += 256) cz[i] = 0u;  // zero halos
  }
  __syncthreads();

  // emb[s*256+e] = sum_l gold_table[gs[s,l], e]   (thread tid = channel e)
  #pragma unroll
  for (int s = 0; s < S_DIM; ++s) {
    float acc = 0.f;
    for (int l = 0; l < 16; ++l) acc += gold_table[gs_s[s*16 + l] * E_DIM + tid];
    emb_s[s * E_DIM + tid] = acc;
  }

  // level-0 activation: l_emb[e,h,w] = sum_k map_table[lmk, e] (row 0 zeroed)
  for (int pos = 0; pos < 16; ++pos) {
    float acc = 0.f;
    for (int k = 0; k < 8; ++k) {
      int idx = lmk_s[pos * 8 + k];
      if (idx != 0) acc += map_table[idx * E_DIM + tid];
    }
    curpad[0][(((pos >> 2) + 1) * 6 + ((pos & 3) + 1)) * E_DIM + tid] = (__bf16)acc;
  }
  __syncthreads();

  const int wave = tid >> 5, lane = tid & 31;
  const int pcol = lane & 15;                 // pixel (B/D column)
  const int ph = pcol >> 2, pw = pcol & 3;
  const int hiK  = (lane >> 4) * 16;          // lanes 16-31 carry K 16..31
  const int mt0  = wave * 2, mt1 = wave * 2 + 1;
  const int rofs = (lane < 16) ? 0 : 8;       // D row offset per ISA C/D layout

  for (int j = 0; j < 3; ++j) {
    const int rbuf = j & 1, wbuf = rbuf ^ 1;
    accum_logits(curpad[rbuf], emb_s, logits_s, j, tid);

    const int a = actions[b * 3 + j];         // uniform -> scalar load
    const __bf16* A0 = Apack + ((size_t)(a * 16 + mt0) * KT_N) * FRAG_ELEMS + lane * 16;
    const __bf16* A1 = Apack + ((size_t)(a * 16 + mt1) * KT_N) * FRAG_ELEMS + lane * 16;

    v8f acc0 = {0.f,0.f,0.f,0.f,0.f,0.f,0.f,0.f};
    v8f acc1 = acc0;

    for (int kt = 0; kt < KT_N; ++kt) {
      const int t  = kt >> 3;
      const int dh = t / 3, dw = t - dh * 3;
      const int ibase = (kt & 7) * 32;
      // B fragment: 16 consecutive channels at one spatial tap -> 32B ds load
      const __bf16* bp = &curpad[rbuf][((ph + dh) * 6 + (pw + dw)) * E_DIM + ibase + hiK];
      v16bf bfrag = *(const v16bf*)bp;

      FragA fa0, fa1;
      const uint4* p0 = (const uint4*)(A0 + (size_t)kt * FRAG_ELEMS);
      const uint4* p1 = (const uint4*)(A1 + (size_t)kt * FRAG_ELEMS);
      fa0.u[0] = p0[0]; fa0.u[1] = p0[1];
      fa1.u[0] = p1[0]; fa1.u[1] = p1[1];

      acc0 = __builtin_amdgcn_wmma_f32_16x16x32_bf16(false, fa0.v, false, bfrag,
                                                     (short)0, acc0, false, false);
      acc1 = __builtin_amdgcn_wmma_f32_16x16x32_bf16(false, fa1.v, false, bfrag,
                                                     (short)0, acc1, false, false);
    }

    // D: vgpr r, lanes0-15 -> row r, lanes16-31 -> row r+8; column = pixel
    __bf16* op = &curpad[wbuf][((ph + 1) * 6 + (pw + 1)) * E_DIM];
    #pragma unroll
    for (int r = 0; r < 8; ++r) {
      op[mt0 * 16 + r + rofs] = (__bf16)acc0[r];
      op[mt1 * 16 + r + rofs] = (__bf16)acc1[r];
    }
    __syncthreads();
  }

  accum_logits(curpad[1], emb_s, logits_s, 3, tid);
  __syncthreads();

  if (tid == 0) {
    float l[16], pr[16];
    float mx = -3.4e38f;
    for (int i = 0; i < 16; ++i) { l[i] = logits_s[i]; mx = fmaxf(mx, l[i]); }
    float s = 0.f;
    for (int i = 0; i < 16; ++i) { pr[i] = __expf(l[i] - mx); s += pr[i]; }
    float inv = 1.f / s;
    for (int i = 0; i < 16; ++i) { pr[i] *= inv; prob_out[b * 16 + i] = pr[i]; }
    // reference applies log_softmax to prob (double softmax) -> replicate
    float mx2 = -3.4e38f;
    for (int i = 0; i < 16; ++i) mx2 = fmaxf(mx2, pr[i]);
    float s2 = 0.f;
    for (int i = 0; i < 16; ++i) s2 += __expf(pr[i] - mx2);
    float lse = mx2 + __logf(s2);
    int yt = yv[b * 2] * 4 + yv[b * 2 + 1];
    loss_b[b] = -(pr[yt] - lse);
    int am = 0; float bv = pr[0];
    for (int i = 1; i < 16; ++i) if (pr[i] > bv) { bv = pr[i]; am = i; }
    acc_b[b] = (am == yt) ? 1.0f : 0.0f;
  }
}

// ---------------- deterministic final reduction -----------------------------
__global__ void __launch_bounds__(256)
reduce_kernel(const float* __restrict__ loss_b, const float* __restrict__ acc_b,
              float* __restrict__ out, int n)
{
  __shared__ float sl[256], sa[256];
  float l = 0.f, a = 0.f;
  for (int i = threadIdx.x; i < n; i += 256) { l += loss_b[i]; a += acc_b[i]; }
  sl[threadIdx.x] = l; sa[threadIdx.x] = a;
  __syncthreads();
  for (int s = 128; s > 0; s >>= 1) {
    if (threadIdx.x < (unsigned)s) {
      sl[threadIdx.x] += sl[threadIdx.x + s];
      sa[threadIdx.x] += sa[threadIdx.x + s];
    }
    __syncthreads();
  }
  if (threadIdx.x == 0) {
    out[0] = sl[0] / (float)n;   // loss mean
    out[1] = sa[0] / (float)n;   // acc mean
  }
}

// ---------------------------------------------------------------------------
extern "C" void kernel_launch(void* const* d_in, const int* in_sizes, int n_in,
                              void* d_out, int out_size, void* d_ws, size_t ws_size,
                              hipStream_t stream)
{
  const float* gold_table   = (const float*)d_in[0];   // (11,256)
  const float* map_table    = (const float*)d_in[1];   // (128,256)
  const float* conv_weight  = (const float*)d_in[2];   // (256,256,3,3)
  const float* action_table = (const float*)d_in[3];   // (4,32)
  const float* action_W     = (const float*)d_in[4];   // (9,32)
  const float* action_b     = (const float*)d_in[5];   // (9,)
  const int*   goldstandard = (const int*)d_in[6];     // (B,4,16)
  const int*   actions      = (const int*)d_in[7];     // (B,3)
  const int*   landmarks    = (const int*)d_in[8];     // (B,4,4,8)
  const int*   yv           = (const int*)d_in[9];     // (B,2)
  float* out = (float*)d_out;                          // [loss, acc, prob(B*16)]

  const int B = in_sizes[9] / 2;                       // 2048

  char* ws = (char*)d_ws;
  __bf16* Apack  = (__bf16*)ws;                        // 4.5 MiB swizzled weights
  float*  loss_b = (float*)(ws + APACK_BYTES);
  float*  acc_b  = loss_b + B;

  // 4 actions * 16 M-tiles * 72 K-tiles fragments, 32 lanes each
  pack_weights_kernel<<<(4 * 16 * KT_N * 32) / 256, 256, 0, stream>>>(
      conv_weight, action_table, action_W, action_b, Apack);

  location_predictor_kernel<<<B, 256, 0, stream>>>(
      gold_table, map_table, goldstandard, actions, landmarks, yv,
      Apack, out + 2, loss_b, acc_b);

  reduce_kernel<<<1, 256, 0, stream>>>(loss_b, acc_b, out, B);
}
